// ImageEnergyTransformer_206158430753
// MI455X (gfx1250) — matrix operand order repaired and checked by reading, not compiled, not run on hardware
//
#include <hip/hip_runtime.h>

typedef __attribute__((ext_vector_type(16))) _Float16 v16h;
typedef __attribute__((ext_vector_type(8)))  _Float16 v8h;
typedef __attribute__((ext_vector_type(8)))  float    v8f;
typedef int v4i_async __attribute__((vector_size(16)));

// Optional CDNA5 async global->LDS path (probe via __has_builtin; falls back
// to synchronous b128 copies if the toolchain doesn't declare the builtins).
#if defined(__has_builtin)
#  if __has_builtin(__builtin_amdgcn_global_load_async_to_lds_b128) && \
      __has_builtin(__builtin_amdgcn_s_wait_asynccnt)
#    define HAVE_ASYNC_LDS 1
#  endif
#endif
#ifndef HAVE_ASYNC_LDS
#  define HAVE_ASYNC_LDS 0
#endif

__device__ __forceinline__ void stage16B(const _Float16* g, _Float16* l)
{
#if HAVE_ASYNC_LDS
    __builtin_amdgcn_global_load_async_to_lds_b128(
        (__attribute__((address_space(1))) v4i_async*)g,
        (__attribute__((address_space(3))) v4i_async*)l, 0, 0);
#else
    *(v8h*)l = *(const v8h*)g;
#endif
}

__device__ __forceinline__ void wait_async_stage()
{
#if HAVE_ASYNC_LDS
    __builtin_amdgcn_s_wait_asynccnt(0);
#endif
}

// ---------------- problem constants ----------------
#define CC      3
#define HH      224
#define WWID    224
#define PP      16
#define KHW     14
#define NPATCH  196
#define DD      768           // model dim == PATCH_ELEMS
#define NHEADS  12
#define YY      64
#define MHID    3072
#define NMASK   100
#define NSTEPS  12
#define NTOK    197
#define BB      16
#define ALPHA_F 0.1f
#define BETA_F  0.125f        // 1/sqrt(64)
#define ROW_E   (NTOK*DD)     // 151296
#define ATT_LD  208           // padded attention row stride (16B-aligned f16 rows)
#define ATT_Z   (NTOK*ATT_LD) // 40976, per-(b,h) slice

// =====================================================================
// Generic WMMA GEMM:  D[z] = op( alpha * A[z]*B[z] (+bias) (+=C) (relu) )
// A is MxK via (ars,acs) strides; B is KxN via (brs,bcs); z offsets are
// split as (z/zdiv)*zo + (z%zdiv)*zi to address [batch][head] layouts.
// Block: 128 threads (4 wave32). Block tile 128x64, each wave 32x64
// (2 A-frags x 4 B-frags = 8 v_wmma per 32-K chunk). K staged in LDS.
// Staging: async/b128 fast paths + branch-free clamped slow path.
// =====================================================================
__global__ __launch_bounds__(128) void gemm_wmma_kernel(
    const _Float16* __restrict__ A, long ars, long acs, long azo, long azi, int azdiv,
    const _Float16* __restrict__ Bm, long brs, long bcs, long bzo, long bzi, int bzdiv,
    float* __restrict__ C, _Float16* __restrict__ C16,
    long crs, long czo, long czi, int czdiv,
    const float* __restrict__ bias,
    int Mdim, int Ndim, int Kdim,
    float alpha, int accum, int relu)
{
    __shared__ __align__(16) _Float16 As[128][40];   // [m][k], 80B rows -> 16B aligned
    __shared__ __align__(16) _Float16 BsT[64][40];   // [n][k]

    const int tid  = threadIdx.x;
    const int lane = tid & 31;
    const int w    = tid >> 5;
    const int m0   = blockIdx.y << 7;
    const int n0   = blockIdx.x << 6;
    const int z    = blockIdx.z;

    const long az = (long)(z / azdiv) * azo + (long)(z % azdiv) * azi;
    const long bz = (long)(z / bzdiv) * bzo + (long)(z % bzdiv) * bzi;
    const long cz = (long)(z / czdiv) * czo + (long)(z % czdiv) * czi;

    const bool a_vec_k = (acs == 1) && ((ars & 7) == 0) && ((az & 7) == 0);
    const bool a_vec_m = (ars == 1) && ((acs & 7) == 0) && ((az & 7) == 0);
    const bool b_vec_k = (brs == 1) && ((bcs & 7) == 0) && ((bz & 7) == 0);
    const bool b_vec_n = (bcs == 1) && ((brs & 7) == 0) && ((bz & 7) == 0);

    v8f acc[2][4] = {};

    for (int kt = 0; kt < Kdim; kt += 32) {
        const bool kfull = (kt + 32 <= Kdim);

        // ---------------- stage A chunk (128 x 32) ----------------
        if (a_vec_k && kfull) {
            // K-contiguous rows: 16B copies straight into LDS (async if available)
            for (int s = tid; s < 512; s += 128) {
                int r = s >> 2, kg = (s & 3) << 3;
                int row = m0 + r; if (row >= Mdim) row = Mdim - 1;   // dup rows are discarded later
                stage16B(A + az + (long)row * ars + (kt + kg), &As[r][kg]);
            }
        } else if (a_vec_m && kfull && (m0 + 128 <= Mdim)) {
            // M-contiguous (transposed A): b128 along M, scatter to LDS
            for (int s = tid; s < 512; s += 128) {
                int rg = (s >> 5) << 3, c = s & 31;
                v8h v = *(const v8h*)(A + az + (long)(m0 + rg) + (long)(kt + c) * acs);
#pragma unroll
                for (int e = 0; e < 8; ++e) As[rg + e][c] = v[e];
            }
        } else {
            // branch-free clamped scalar path
            for (int i = tid; i < 128 * 32; i += 128) {
                int r = i >> 5, c = i & 31;
                int row = m0 + r, col = kt + c;
                int rr = row < Mdim ? row : Mdim - 1;
                int cc = col < Kdim ? col : Kdim - 1;
                _Float16 v = A[az + (long)rr * ars + (long)cc * acs];
                if (row >= Mdim || col >= Kdim) v = (_Float16)0.0f;
                As[r][c] = v;
            }
        }

        // ---------------- stage B chunk (32 x 64) into BsT[n][k] ----------------
        if (b_vec_k && kfull) {
            for (int s = tid; s < 256; s += 128) {
                int n = s >> 2, kg = (s & 3) << 3;
                int col = n0 + n; if (col >= Ndim) col = Ndim - 1;   // dup cols discarded later
                stage16B(Bm + bz + (kt + kg) + (long)col * bcs, &BsT[n][kg]);
            }
        } else if (b_vec_n && kfull && (n0 + 64 <= Ndim)) {
            for (int s = tid; s < 256; s += 128) {
                int k = s >> 3, ng = (s & 7) << 3;
                v8h v = *(const v8h*)(Bm + bz + (long)(kt + k) * brs + (n0 + ng));
#pragma unroll
                for (int e = 0; e < 8; ++e) BsT[ng + e][k] = v[e];
            }
        } else {
            for (int i = tid; i < 64 * 32; i += 128) {
                int n = i & 63, kk = i >> 6;
                int row = kt + kk, col = n0 + n;
                int rr = row < Kdim ? row : Kdim - 1;
                int cc = col < Ndim ? col : Ndim - 1;
                _Float16 v = Bm[bz + (long)rr * brs + (long)cc * bcs];
                if (row >= Kdim || col >= Ndim) v = (_Float16)0.0f;
                BsT[n][kk] = v;
            }
        }
        wait_async_stage();     // drain ASYNCcnt (no-op if async path unused)
        __syncthreads();

        // ---------------- fragments + 8 WMMAs ----------------
        const int kb  = (lane >> 4) << 3;     // A: 0 or 8
    const int kb2 = (lane >> 4) << 3;     // B: 0 or 8
        v16h afrag[2];
#pragma unroll
        for (int mi = 0; mi < 2; ++mi) {
            int arow = (w << 5) + (mi << 4) + (lane & 15);
            v8h alo = *(const v8h*)&As[arow][kb];
            v8h ahi = *(const v8h*)&As[arow][kb + 16];
            afrag[mi] = __builtin_shufflevector(alo, ahi,
                          0,1,2,3,4,5,6,7,8,9,10,11,12,13,14,15);
        }
#pragma unroll
        for (int j = 0; j < 4; ++j) {
            int bcol = (j << 4) + (lane & 15);
            v8h b0 = *(const v8h*)&BsT[bcol][kb2];
            v8h b1 = *(const v8h*)&BsT[bcol][kb2 + 16];
            v16h bf = __builtin_shufflevector(b0, b1,
                        0,1,2,3,4,5,6,7,8,9,10,11,12,13,14,15);
            acc[0][j] = __builtin_amdgcn_wmma_f32_16x16x32_f16(
                            false, afrag[0], false, bf, (short)0, acc[0][j], false, false);
            acc[1][j] = __builtin_amdgcn_wmma_f32_16x16x32_f16(
                            false, afrag[1], false, bf, (short)0, acc[1][j], false, false);
        }
        __syncthreads();
    }

    // ---------------- epilogue ----------------
    const int cbase = n0 + (lane & 15);
#pragma unroll
    for (int mi = 0; mi < 2; ++mi) {
        const int rbase = m0 + (w << 5) + (mi << 4) + ((lane >> 4) << 3);
#pragma unroll
        for (int j = 0; j < 4; ++j) {
            int col = cbase + (j << 4);
            if (col >= Ndim) continue;
#pragma unroll
            for (int e = 0; e < 8; ++e) {
                int row = rbase + e;
                if (row >= Mdim) continue;
                float v = alpha * acc[mi][j][e];
                if (bias) v += bias[col];
                long coff = cz + (long)row * crs + col;
                if (accum && C) v += C[coff];
                if (relu) v = fmaxf(v, 0.0f);
                if (C)   C[coff]   = v;
                if (C16) C16[coff] = (_Float16)v;
            }
        }
    }
}

// ---------------- helper kernels ----------------
__global__ void cvt_f16_kernel(const float* __restrict__ s, _Float16* __restrict__ d, long n)
{
    for (long i = (long)blockIdx.x * blockDim.x + threadIdx.x; i < n;
         i += (long)gridDim.x * blockDim.x)
        d[i] = (_Float16)s[i];
}

// Wq (H,D,Y) -> packed [d][h*64+y] and transposed [h*64+y][d]
__global__ void cvt_qk_kernel(const float* __restrict__ W, _Float16* __restrict__ Wp,
                              _Float16* __restrict__ WpT)
{
    const long total = (long)NHEADS * DD * YY;
    for (long i = (long)blockIdx.x * blockDim.x + threadIdx.x; i < total;
         i += (long)gridDim.x * blockDim.x) {
        int h = (int)(i / (DD * YY));
        int r = (int)(i % (DD * YY));
        int d = r / YY, y = r % YY;
        _Float16 v = (_Float16)W[i];
        Wp[(long)d * DD + h * YY + y]    = v;
        WpT[((long)h * YY + y) * DD + d] = v;
    }
}

__global__ void cvt_xi_kernel(const float* __restrict__ Xi, _Float16* __restrict__ Xi16,
                              _Float16* __restrict__ XiT16)
{
    const long total = (long)DD * MHID;
    for (long i = (long)blockIdx.x * blockDim.x + threadIdx.x; i < total;
         i += (long)gridDim.x * blockDim.x) {
        int d = (int)(i / MHID), m = (int)(i % MHID);
        _Float16 v = (_Float16)Xi[i];
        Xi16[i]                 = v;
        XiT16[(long)m * DD + d] = v;
    }
}

__global__ void patchify_kernel(const float* __restrict__ img, _Float16* __restrict__ pt)
{
    const long total = (long)BB * NPATCH * DD;
    for (long i = (long)blockIdx.x * blockDim.x + threadIdx.x; i < total;
         i += (long)gridDim.x * blockDim.x) {
        int b  = (int)(i / ((long)NPATCH * DD));
        int r  = (int)(i % ((long)NPATCH * DD));
        int n  = r / DD, e = r % DD;
        int kh = n / KHW, kw = n % KHW;
        int c  = e >> 8; int r2 = e & 255;
        int py = r2 >> 4, px = r2 & 15;
        long src = (((long)b * CC + c) * HH + (kh * PP + py)) * WWID + (kw * PP + px);
        pt[i] = (_Float16)img[src];
    }
}

// mask-token substitution (exact jnp.nonzero(size=100, fill=0) semantics),
// cls token, encoder bias, positional embedding
__global__ void finalize_kernel(float* __restrict__ x, const int* __restrict__ mask,
                                const float* __restrict__ encB, const float* __restrict__ cls,
                                const float* __restrict__ mtok, const float* __restrict__ pos)
{
    __shared__ unsigned char repl[NPATCH];
    const int b = blockIdx.x;
    if (threadIdx.x == 0) {
        int r = 0;
        for (int n = 0; n < NPATCH; ++n) {
            int m = mask[b * NPATCH + n];
            repl[n] = (m == 1 && r < NMASK) ? 1 : 0;
            if (m == 1) r++;
        }
        if (r < NMASK) repl[0] = 1;   // fill_value=0 overwrites token 0
    }
    __syncthreads();
    float* xb = x + (long)b * ROW_E;
    for (int i = threadIdx.x; i < ROW_E; i += blockDim.x) {
        int t = i / DD, d = i % DD;
        float v;
        if (t == 0)            v = cls[d];
        else if (repl[t - 1])  v = mtok[d];
        else                   v = xb[i] + encB[d];
        xb[i] = v + pos[i];
    }
}

__global__ void lnorm_kernel(const float* __restrict__ x, _Float16* __restrict__ g,
                             const float* __restrict__ gamma, const float* __restrict__ delta)
{
    __shared__ float red[256];
    const int tid = threadIdx.x;
    const float* xr = x + (long)blockIdx.x * DD;
    float s = 0.0f;
    for (int d = tid; d < DD; d += 256) s += xr[d];
    red[tid] = s; __syncthreads();
    for (int o = 128; o > 0; o >>= 1) { if (tid < o) red[tid] += red[tid + o]; __syncthreads(); }
    const float mean = red[0] / (float)DD; __syncthreads();
    float s2 = 0.0f;
    for (int d = tid; d < DD; d += 256) { float t = xr[d] - mean; s2 += t * t; }
    red[tid] = s2; __syncthreads();
    for (int o = 128; o > 0; o >>= 1) { if (tid < o) red[tid] += red[tid + o]; __syncthreads(); }
    const float rstd = rsqrtf(red[0] / (float)DD + 1e-5f);
    const float gm = gamma[0];
    _Float16* gr = g + (long)blockIdx.x * DD;
    for (int d = tid; d < DD; d += 256)
        gr[d] = (_Float16)(gm * (xr[d] - mean) * rstd + delta[d]);
}

// softmax over padded rows (stride ATT_LD); zeroes pad tail of s16
__global__ void softmax_kernel(const float* __restrict__ att, _Float16* __restrict__ s16)
{
    __shared__ float red[256];
    const int tid = threadIdx.x;
    const float* ar = att + (long)blockIdx.x * ATT_LD;
    _Float16*    sr = s16 + (long)blockIdx.x * ATT_LD;
    float m = -3.4e38f;
    for (int i = tid; i < NTOK; i += 256) m = fmaxf(m, ar[i]);
    red[tid] = m; __syncthreads();
    for (int o = 128; o > 0; o >>= 1) { if (tid < o) red[tid] = fmaxf(red[tid], red[tid + o]); __syncthreads(); }
    const float amax = red[0]; __syncthreads();
    float s = 0.0f;
    for (int i = tid; i < NTOK; i += 256) s += __expf(BETA_F * (ar[i] - amax));
    red[tid] = s; __syncthreads();
    for (int o = 128; o > 0; o >>= 1) { if (tid < o) red[tid] += red[tid + o]; __syncthreads(); }
    const float inv = 1.0f / red[0];
    for (int i = tid; i < NTOK; i += 256)
        sr[i] = (_Float16)(__expf(BETA_F * (ar[i] - amax)) * inv);
    if (tid < ATT_LD - NTOK) sr[NTOK + tid] = (_Float16)0.0f;
}

__global__ void unpatchify_kernel(const float* __restrict__ dec, float* __restrict__ out)
{
    const long total = (long)BB * CC * HH * WWID;
    for (long i = (long)blockIdx.x * blockDim.x + threadIdx.x; i < total;
         i += (long)gridDim.x * blockDim.x) {
        int b  = (int)(i / ((long)CC * HH * WWID));
        int r  = (int)(i % ((long)CC * HH * WWID));
        int c  = r / (HH * WWID);
        int r2 = r % (HH * WWID);
        int h  = r2 / WWID, ww = r2 % WWID;
        int kh = h >> 4, py = h & 15, kw = ww >> 4, px = ww & 15;
        int n  = kh * KHW + kw;
        int e  = (c << 8) + (py << 4) + px;
        out[i] = dec[((long)b * NPATCH + n) * DD + e];
    }
}

// ---------------- host side ----------------
static inline int cdiv_i(long a, int b) { return (int)((a + b - 1) / b); }

static void launch_gemm(hipStream_t stream,
    const _Float16* A, long ars, long acs, long azo, long azi, int azdiv,
    const _Float16* Bm, long brs, long bcs, long bzo, long bzi, int bzdiv,
    float* C, _Float16* C16, long crs, long czo, long czi, int czdiv,
    const float* bias, int M, int N, int K, int Z,
    float alpha, int accum, int relu)
{
    dim3 grid((N + 63) / 64, (M + 127) / 128, Z);
    gemm_wmma_kernel<<<grid, 128, 0, stream>>>(
        A, ars, acs, azo, azi, azdiv,
        Bm, brs, bcs, bzo, bzi, bzdiv,
        C, C16, crs, czo, czi, czdiv,
        bias, M, N, K, alpha, accum, relu);
}

extern "C" void kernel_launch(void* const* d_in, const int* in_sizes, int n_in,
                              void* d_out, int out_size, void* d_ws, size_t ws_size,
                              hipStream_t stream)
{
    (void)in_sizes; (void)n_in; (void)out_size; (void)ws_size;
    const float* img   = (const float*)d_in[0];
    const int*   mask  = (const int*)  d_in[1];
    const float* encW  = (const float*)d_in[2];
    const float* encB  = (const float*)d_in[3];
    const float* decW  = (const float*)d_in[4];
    const float* decB  = (const float*)d_in[5];
    const float* cls   = (const float*)d_in[6];
    const float* mtok  = (const float*)d_in[7];
    const float* pos   = (const float*)d_in[8];
    const float* Wq    = (const float*)d_in[9];
    const float* Wk    = (const float*)d_in[10];
    const float* Xi    = (const float*)d_in[11];
    const float* gamma = (const float*)d_in[12];
    const float* delta = (const float*)d_in[13];
    float* out = (float*)d_out;

    size_t off = 0;
    auto take = [&](size_t bytes) -> char* {
        char* p = (char*)d_ws + off;
        off += (bytes + 255) & ~(size_t)255;
        return p;
    };

    float*    xws    = (float*)   take((size_t)BB * ROW_E * 4);
    float*    att    = (float*)   take((size_t)BB * NHEADS * ATT_Z * 4);
    _Float16* g16    = (_Float16*)take((size_t)BB * ROW_E * 2);
    _Float16* q16    = (_Float16*)take((size_t)BB * ROW_E * 2);
    _Float16* k16    = (_Float16*)take((size_t)BB * ROW_E * 2);
    _Float16* s16    = (_Float16*)take((size_t)BB * NHEADS * ATT_Z * 2);
    _Float16* u16    = (_Float16*)take((size_t)BB * ROW_E * 2);
    _Float16* v16    = (_Float16*)take((size_t)BB * ROW_E * 2);
    _Float16* hid16  = (_Float16*)take((size_t)BB * NTOK * MHID * 2);
    _Float16* pt16   = (_Float16*)take((size_t)BB * NPATCH * DD * 2);
    _Float16* encW16 = (_Float16*)take((size_t)DD * DD * 2);
    _Float16* decW16 = (_Float16*)take((size_t)DD * DD * 2);
    _Float16* Wq16   = (_Float16*)take((size_t)DD * DD * 2);
    _Float16* WqT16  = (_Float16*)take((size_t)DD * DD * 2);
    _Float16* Wk16   = (_Float16*)take((size_t)DD * DD * 2);
    _Float16* WkT16  = (_Float16*)take((size_t)DD * DD * 2);
    _Float16* Xi16   = (_Float16*)take((size_t)DD * MHID * 2);
    _Float16* XiT16  = (_Float16*)take((size_t)DD * MHID * 2);
    float* dec32 = att;   // decoder output aliases attention scratch (free by then)

    // ---- weight conversion (L2-resident f16 copies, incl. transposes) ----
    cvt_f16_kernel<<<cdiv_i((long)DD * DD, 256), 256, 0, stream>>>(encW, encW16, (long)DD * DD);
    cvt_f16_kernel<<<cdiv_i((long)DD * DD, 256), 256, 0, stream>>>(decW, decW16, (long)DD * DD);
    cvt_qk_kernel<<<cdiv_i((long)NHEADS * DD * YY, 256), 256, 0, stream>>>(Wq, Wq16, WqT16);
    cvt_qk_kernel<<<cdiv_i((long)NHEADS * DD * YY, 256), 256, 0, stream>>>(Wk, Wk16, WkT16);
    cvt_xi_kernel<<<cdiv_i((long)DD * MHID, 256), 256, 0, stream>>>(Xi, Xi16, XiT16);

    // ---- encoder: tok = patchify(img) @ enc_W  (bias/mask/pos in finalize) ----
    patchify_kernel<<<cdiv_i((long)BB * NPATCH * DD, 256), 256, 0, stream>>>(img, pt16);
    launch_gemm(stream, pt16, DD, 1, (long)NPATCH * DD, 0, 1,
                encW16, DD, 1, 0, 0, 1,
                xws + DD, nullptr, DD, ROW_E, 0, 1,
                nullptr, NPATCH, DD, DD, BB, 1.0f, 0, 0);
    finalize_kernel<<<BB, 256, 0, stream>>>(xws, mask, encB, cls, mtok, pos);

    // ---- 12 energy-descent steps ----
    for (int step = 0; step < NSTEPS; ++step) {
        lnorm_kernel<<<BB * NTOK, 256, 0, stream>>>(xws, g16, gamma, delta);
        // q = g @ Wq, k = g @ Wk  (packed per-head columns)
        launch_gemm(stream, g16, DD, 1, ROW_E, 0, 1, Wq16, DD, 1, 0, 0, 1,
                    nullptr, q16, DD, ROW_E, 0, 1, nullptr, NTOK, DD, DD, BB, 1.0f, 0, 0);
        launch_gemm(stream, g16, DD, 1, ROW_E, 0, 1, Wk16, DD, 1, 0, 0, 1,
                    nullptr, k16, DD, ROW_E, 0, 1, nullptr, NTOK, DD, DD, BB, 1.0f, 0, 0);
        // A[b,h] = q k^T   (z = b*12+h), padded row stride
        launch_gemm(stream, q16, DD, 1, ROW_E, YY, NHEADS,
                    k16, 1, DD, ROW_E, YY, NHEADS,
                    att, nullptr, ATT_LD, (long)ATT_Z, 0, 1,
                    nullptr, NTOK, NTOK, YY, BB * NHEADS, 1.0f, 0, 0);
        softmax_kernel<<<BB * NHEADS * NTOK, 256, 0, stream>>>(att, s16);
        // u = s @ k ; v = s^T @ q   (per head)
        launch_gemm(stream, s16, ATT_LD, 1, (long)ATT_Z, 0, 1,
                    k16, DD, 1, ROW_E, YY, NHEADS,
                    nullptr, u16, DD, ROW_E, YY, NHEADS,
                    nullptr, NTOK, YY, NTOK, BB * NHEADS, 1.0f, 0, 0);
        launch_gemm(stream, s16, 1, ATT_LD, (long)ATT_Z, 0, 1,
                    q16, DD, 1, ROW_E, YY, NHEADS,
                    nullptr, v16, DD, ROW_E, YY, NHEADS,
                    nullptr, NTOK, YY, NTOK, BB * NHEADS, 1.0f, 0, 0);
        // hid = relu(g @ Xi)
        launch_gemm(stream, g16, DD, 1, ROW_E, 0, 1, Xi16, MHID, 1, 0, 0, 1,
                    nullptr, hid16, MHID, (long)NTOK * MHID, 0, 1,
                    nullptr, NTOK, MHID, DD, BB, 1.0f, 0, 1);
        // x += ALPHA * (u @ Wq^T + v @ Wk^T + hid @ Xi^T)   (== x - ALPHA*grad)
        launch_gemm(stream, u16, DD, 1, ROW_E, 0, 1, WqT16, DD, 1, 0, 0, 1,
                    xws, nullptr, DD, ROW_E, 0, 1, nullptr, NTOK, DD, DD, BB, ALPHA_F, 1, 0);
        launch_gemm(stream, v16, DD, 1, ROW_E, 0, 1, WkT16, DD, 1, 0, 0, 1,
                    xws, nullptr, DD, ROW_E, 0, 1, nullptr, NTOK, DD, DD, BB, ALPHA_F, 1, 0);
        launch_gemm(stream, hid16, MHID, 1, (long)NTOK * MHID, 0, 1, XiT16, DD, 1, 0, 0, 1,
                    xws, nullptr, DD, ROW_E, 0, 1, nullptr, NTOK, DD, MHID, BB, ALPHA_F, 1, 0);
    }

    // ---- decode ----
    lnorm_kernel<<<BB * NTOK, 256, 0, stream>>>(xws, g16, gamma, delta);
    launch_gemm(stream, g16 + DD, DD, 1, ROW_E, 0, 1,      // skip cls row
                decW16, DD, 1, 0, 0, 1,
                dec32, nullptr, DD, (long)NPATCH * DD, 0, 1,
                decB, NPATCH, DD, DD, BB, 1.0f, 0, 0);
    unpatchify_kernel<<<cdiv_i((long)BB * CC * HH * WWID, 256), 256, 0, stream>>>(dec32, out);
}
